// TtMambaSSM_72052371358161
// MI455X (gfx1250) — matrix operand order repaired and checked by reading, not compiled, not run on hardware
//
#include <hip/hip_runtime.h>

// ---------------------------------------------------------------------------
// Mamba SSM for gfx1250 (MI455X).
//   Weights pre-transposed to [N][K] bf16 at convert time, so every WMMA
//   operand slab is K-contiguous:
//     Big GEMMs : 128x128 macro-tile, double-buffered LDS; A AND B staged via
//                 global_load_async_to_lds_b128 (ASYNCcnt) overlapped with
//                 compute; fragments via ds_load_b128 (80B-padded rows);
//                 V_WMMA_F32_16X16X32_BF16, f32 accumulate, templated epilogue.
//     B/C proj  : fused kernel, one wave per token tile, contiguous b128
//                 fragment loads, two accumulators (Bp, Cp).
//   Scan        : fused discretize + recurrence + einsum, h[16] in registers.
// ---------------------------------------------------------------------------

typedef __attribute__((ext_vector_type(16))) __bf16 v16bf;
typedef __attribute__((ext_vector_type(8)))  float  v8f;

#define T_TOK 2048   // B*L tokens
#define DIN   2048
#define RNK   128
#define NST   16
#define LSEQ  1024
#define NB    2

#define BM 128
#define BN 128
#define BK 32
#define LSTRIDE 40   // LDS row stride in elements (80 B, 16B-aligned, padded)

// ---- f32 -> bf16 conversion ----------------------------------------------
__global__ void cvt_f32_bf16(const float* __restrict__ in,
                             __bf16* __restrict__ out, int n) {
  int i = blockIdx.x * blockDim.x + threadIdx.x;
  if (i < n) out[i] = (__bf16)in[i];
}

// f32 [R][C] -> bf16 [C][R] (weights become K-contiguous for WMMA B frags)
__global__ void cvt_f32_bf16_T(const float* __restrict__ in,
                               __bf16* __restrict__ out, int R, int C) {
  int i = blockIdx.x * blockDim.x + threadIdx.x;
  if (i < R * C) {
    int r = i / C, c = i % C;
    out[(size_t)c * R + r] = (__bf16)in[i];
  }
}

// branchless thresholded softplus: v_cndmask select, no EXEC games
__device__ __forceinline__ float softplus_thresh(float v) {
  float sp = __logf(1.0f + __expf(fminf(v, 20.0f)));
  return (v > 20.0f) ? v : sp;
}

// ---- macro-tiled, double-buffered WMMA bf16 GEMM --------------------------
// A: [M][K] row-major bf16.  Bt: [N][K] row-major bf16 (pre-transposed).
// M, N multiples of 128; K multiple of 32.
template <bool HAS_BIAS, bool SOFTPLUS, bool OUT_BF16>
__global__ void __launch_bounds__(256)
wmma_gemm_bf16_tiled(const __bf16* __restrict__ A,
                     const __bf16* __restrict__ Bt,
                     void* __restrict__ Dst,
                     const float* __restrict__ bias,
                     int M, int N, int K) {
  __shared__ __bf16 Asl[2][BM * LSTRIDE];     // 2 x 10 KB
  __shared__ __bf16 Bsl[2][BN * LSTRIDE];     // 2 x 10 KB

  const int tid  = threadIdx.x;
  const int lane = tid & 31;
  const int wv   = tid >> 5;                  // wave 0..7 -> M-subtile
  const int m0   = blockIdx.y * BM;
  const int n0   = blockIdx.x * BN;
  const int row  = lane & 15;
  const int hi   = lane >> 4;                 // half-wave select

  // async-DMA one 128x32 bf16 slab (512 x 16B chunks, 2 per thread)
  auto stage_slab = [&](const __bf16* src, int row0, const __bf16* lds,
                        int k0) {
    const unsigned lbase = (unsigned)(uintptr_t)lds;
    int chunk = tid;
#pragma unroll
    for (int r = 0; r < 2; ++r, chunk += 256) {
      const int rr = chunk >> 2;              // 0..127
      const int qt = chunk & 3;               // 16 B quarter of 64 B row
      const __bf16* gp = src + (size_t)(row0 + rr) * K + k0 + qt * 8;
      const unsigned laddr = lbase + (unsigned)(rr * (LSTRIDE * 2) + qt * 16);
      asm volatile("global_load_async_to_lds_b128 %0, %1, off"
                   :: "v"(laddr), "v"(gp) : "memory");
    }
  };

  v8f acc[BN / 16];
#pragma unroll
  for (int t = 0; t < BN / 16; ++t) acc[t] = (v8f){};

  const int nk = K / BK;
  stage_slab(A, m0, &Asl[0][0], 0);
  stage_slab(Bt, n0, &Bsl[0][0], 0);
  asm volatile("s_wait_asynccnt 0x0" ::: "memory");
  __syncthreads();

  for (int ks = 0; ks < nk; ++ks) {
    const int cur = ks & 1;
    if (ks + 1 < nk) {                        // overlap DMA with WMMA
      stage_slab(A, m0, &Asl[cur ^ 1][0], (ks + 1) * BK);
      stage_slab(Bt, n0, &Bsl[cur ^ 1][0], (ks + 1) * BK);
    }

    v16bf afrag;
    {
      const __bf16* ap = &Asl[cur][(wv * 16 + row) * LSTRIDE + hi * 16];
#pragma unroll
      for (int j = 0; j < 8; ++j) afrag[j] = ap[j];          // K = hi*16+0..7
#pragma unroll
      for (int j = 0; j < 8; ++j) afrag[8 + j] = ap[8 + j];  //   + 8..15
    }
#pragma unroll
    for (int nt = 0; nt < BN / 16; ++nt) {
      v16bf bfrag;
      const __bf16* bp = &Bsl[cur][(nt * 16 + row) * LSTRIDE + hi * 16];
#pragma unroll
      for (int j = 0; j < 16; ++j) bfrag[j] = bp[j];  // two ds_load_b128
      acc[nt] = __builtin_amdgcn_wmma_f32_16x16x32_bf16(
          false, afrag, false, bfrag, (short)0, acc[nt], false, false);
    }

    asm volatile("s_wait_asynccnt 0x0" ::: "memory");  // next slab landed
    __syncthreads();                                   // cur slab consumed
  }

  // ---- epilogue: C/D layout VGPR v -> row hi*8+v, col lane&15 ------------
  const int cn = lane & 15;
  const int mo = hi * 8;
#pragma unroll
  for (int nt = 0; nt < BN / 16; ++nt) {
    const int gn = n0 + nt * 16 + cn;
    float bv = 0.0f;
    if (HAS_BIAS) bv = bias[gn];               // one load per N-tile
#pragma unroll
    for (int v = 0; v < 8; ++v) {
      float val = acc[nt][v];
      if (HAS_BIAS) val += bv;
      if (SOFTPLUS) val = softplus_thresh(val);
      const size_t o = (size_t)(m0 + wv * 16 + mo + v) * N + gn;
      if (OUT_BF16) ((__bf16*)Dst)[o] = (__bf16)val;
      else          ((float*)Dst)[o]  = val;
    }
  }
}

// ---- fused B/C projection: Bp|Cp[T,16] = x @ {W_B, W_C} -------------------
// WtBC: [32][K] bf16, rows 0..15 = W_B^T, rows 16..31 = W_C^T.
// One wave per 16-token tile; all fragment loads contiguous b128.
__global__ void wmma_gemm_bc(const __bf16* __restrict__ A,
                             const __bf16* __restrict__ WtBC,
                             float* __restrict__ Bp,
                             float* __restrict__ Cp,
                             int M, int K) {
  const int wave = (int)(blockIdx.x * blockDim.x + threadIdx.x) >> 5;
  if (wave >= (M >> 4)) return;            // whole-wave exit, EXEC all-1s
  const int lane = threadIdx.x & 31;
  const int am  = wave << 4;
  const int row = lane & 15;
  const int hi  = lane >> 4;

  v8f accB = {}, accC = {};
  for (int k0 = 0; k0 < K; k0 += 32) {
    v16bf a, bB, bC;
    const __bf16* ap = A + (size_t)(am + row) * K + k0 + hi * 16;
#pragma unroll
    for (int j = 0; j < 8; ++j) a[j] = ap[j];
#pragma unroll
    for (int j = 0; j < 8; ++j) a[8 + j] = ap[8 + j];
    const __bf16* pB = WtBC + (size_t)row * K + k0 + hi * 16;
#pragma unroll
    for (int j = 0; j < 16; ++j) bB[j] = pB[j];
    const __bf16* pC = WtBC + (size_t)(16 + row) * K + k0 + hi * 16;
#pragma unroll
    for (int j = 0; j < 16; ++j) bC[j] = pC[j];
    accB = __builtin_amdgcn_wmma_f32_16x16x32_bf16(
        false, a, false, bB, (short)0, accB, false, false);
    accC = __builtin_amdgcn_wmma_f32_16x16x32_bf16(
        false, a, false, bC, (short)0, accC, false, false);
  }
  const int n = lane & 15;
  const int mo = hi * 8;
#pragma unroll
  for (int v = 0; v < 8; ++v) {
    Bp[(size_t)(am + mo + v) * NST + n] = accB[v];
    Cp[(size_t)(am + mo + v) * NST + n] = accC[v];
  }
}

// ---- fused discretize + scan + einsum -------------------------------------
__global__ void mamba_scan(const float* __restrict__ x,
                           const float* __restrict__ delta,
                           const float* __restrict__ Bp,
                           const float* __restrict__ Cp,
                           const float* __restrict__ A_log,
                           const float* __restrict__ Dpar,
                           float* __restrict__ y) {
  const int idx = blockIdx.x * blockDim.x + threadIdx.x;  // 0 .. NB*DIN-1
  const int b = idx / DIN;
  const int d = idx % DIN;

  float Arow[NST];
#pragma unroll
  for (int n = 0; n < NST; ++n) Arow[n] = -__expf(A_log[d * NST + n]);
  const float Dd = Dpar[d];

  float h[NST];
#pragma unroll
  for (int n = 0; n < NST; ++n) h[n] = 0.0f;

  for (int l = 0; l < LSEQ; ++l) {
    const size_t t = (size_t)b * LSEQ + l;
    const float dv = delta[t * DIN + d];
    const float xv = x[t * DIN + d];
    if (l + 4 < LSEQ) {
      __builtin_prefetch(&delta[(t + 4) * DIN + d], 0, 0);
      __builtin_prefetch(&x[(t + 4) * DIN + d], 0, 0);
    }
    const float dx = dv * xv;
    float accv = 0.0f;
#pragma unroll
    for (int n = 0; n < NST; ++n) {
      const float bb = Bp[t * NST + n];
      const float cc = Cp[t * NST + n];
      h[n] = __expf(dv * Arow[n]) * h[n] + dx * bb;
      accv += h[n] * cc;
    }
    y[t * DIN + d] = accv + xv * Dd;
  }
}

// ---------------------------------------------------------------------------
extern "C" void kernel_launch(void* const* d_in, const int* in_sizes, int n_in,
                              void* d_out, int out_size, void* d_ws, size_t ws_size,
                              hipStream_t stream) {
  const float* x       = (const float*)d_in[0];
  const float* W_delta = (const float*)d_in[1];
  const float* W_dt    = (const float*)d_in[2];
  const float* b_dt    = (const float*)d_in[3];
  const float* W_B     = (const float*)d_in[4];
  const float* W_C     = (const float*)d_in[5];
  const float* A_log   = (const float*)d_in[6];
  const float* Dpar    = (const float*)d_in[7];
  float* y = (float*)d_out;

  char* ws = (char*)d_ws;
  size_t off = 0;
  auto carve = [&](size_t bytes) {
    void* p = ws + off;
    off = (off + bytes + 255) & ~(size_t)255;
    return p;
  };
  __bf16* xb    = (__bf16*)carve((size_t)T_TOK * DIN * 2);  // x bf16 [T][Din]
  __bf16* wdT   = (__bf16*)carve((size_t)RNK * DIN * 2);    // W_delta^T [128][2048]
  __bf16* wdtT  = (__bf16*)carve((size_t)DIN * RNK * 2);    // W_dt^T [2048][128]
  __bf16* wBCt  = (__bf16*)carve((size_t)32 * DIN * 2);     // [W_B^T; W_C^T] [32][2048]
  __bf16* t1b   = (__bf16*)carve((size_t)T_TOK * RNK * 2);  // x@W_delta bf16
  float*  delta = (float*)carve((size_t)T_TOK * DIN * 4);
  float*  Bp    = (float*)carve((size_t)T_TOK * NST * 4);
  float*  Cp    = (float*)carve((size_t)T_TOK * NST * 4);
  (void)ws_size; (void)in_sizes; (void)n_in; (void)out_size;

  cvt_f32_bf16<<<(T_TOK * DIN + 255) / 256, 256, 0, stream>>>(x, xb, T_TOK * DIN);
  cvt_f32_bf16_T<<<(DIN * RNK + 255) / 256, 256, 0, stream>>>(W_delta, wdT, DIN, RNK);
  cvt_f32_bf16_T<<<(RNK * DIN + 255) / 256, 256, 0, stream>>>(W_dt, wdtT, RNK, DIN);
  cvt_f32_bf16_T<<<(DIN * NST + 255) / 256, 256, 0, stream>>>(W_B, wBCt, DIN, NST);
  cvt_f32_bf16_T<<<(DIN * NST + 255) / 256, 256, 0, stream>>>(W_C, wBCt + (size_t)NST * DIN, DIN, NST);

  // GEMM1: t1[T,128] = x @ W_delta  (bf16 out -> feeds GEMM2 directly)
  {
    dim3 grid(RNK / BN, T_TOK / BM);
    wmma_gemm_bf16_tiled<false, false, true><<<grid, 256, 0, stream>>>(
        xb, wdT, t1b, nullptr, T_TOK, RNK, DIN);
  }
  // GEMM2: delta[T,2048] = softplus(t1 @ W_dt + b_dt)  (f32 out)
  {
    dim3 grid(DIN / BN, T_TOK / BM);
    wmma_gemm_bf16_tiled<true, true, false><<<grid, 256, 0, stream>>>(
        t1b, wdtT, delta, b_dt, T_TOK, DIN, RNK);
  }
  // Fused Bp/Cp projection: one pass over x
  {
    int waves = T_TOK / 16;                 // 128 waves
    wmma_gemm_bc<<<(waves * 32 + 255) / 256, 256, 0, stream>>>(
        xb, wBCt, Bp, Cp, T_TOK, DIN);
  }
  // Fused scan + output
  mamba_scan<<<(NB * DIN) / 256, 256, 0, stream>>>(
      x, delta, Bp, Cp, A_log, Dpar, y);
}
